// FullModel_10264971837890
// MI455X (gfx1250) — matrix operand order repaired and checked by reading, not compiled, not run on hardware
//
#include <hip/hip_runtime.h>
#include <hip/hip_bf16.h>

// ---------------------------------------------------------------------------
// Model dims (from reference)
// ---------------------------------------------------------------------------
#define R_   64
#define B_   128
#define IN_  2048
#define HID_ 512
#define ROI_OUT_ 20
#define G1_  64
#define G2_  128
#define FC1_ 512
#define FC2_ 1024
#define OUT_ 6400

typedef __attribute__((ext_vector_type(16))) __bf16 v16bf;
typedef __attribute__((ext_vector_type(8)))  float  v8f;

// Pack two f32 into a dword of two bf16 (RNE): low half = a, high half = b.
__device__ __forceinline__ unsigned int pack_bf16(float a, float b) {
    unsigned int ua = __float_as_uint(a);
    unsigned int ub = __float_as_uint(b);
    ua += 0x7FFFu + ((ua >> 16) & 1u);
    ub += 0x7FFFu + ((ub >> 16) & 1u);
    // result bytes (LSB..MSB) = { ua.b2, ua.b3, ub.b2, ub.b3 }
    return __builtin_amdgcn_perm(ub, ua, 0x07060302u);
}

// ---------------------------------------------------------------------------
// Generic bf16-WMMA GEMM:  C[z] = act(A[z] * B[z] + bias[z])
// A: [M x K] f32 row-major (lda), B: [K x N] f32 row-major (ldb)
// Requirements: M % BM == 0, N % BN == 0, K % 32 == 0.
// Block = 256 threads = 8 waves (WGM x WGN wave grid); BK = 32.
// LDS ping-pong + software pipeline: global loads for slab k+1 are issued
// before the WMMA burst for slab k; pack+LDS-store afterwards.
// ---------------------------------------------------------------------------
template <int BM, int BN, int WGM, int WGN>
__global__ __launch_bounds__(256)
void gemm_bf16_wmma(const float* __restrict__ A, int lda, long long sA,
                    const float* __restrict__ Bm, int ldb, long long sB,
                    const float* __restrict__ bias, long long sBias,
                    float* __restrict__ C, int ldc, long long sC,
                    int K, int relu)
{
    constexpr int MT  = BM / (16 * WGM);   // wmma tiles per wave in M
    constexpr int NT  = BN / (16 * WGN);   // wmma tiles per wave in N
    constexpr int ITA = (BM * 16) / 256;   // A staging dwords per thread
    constexpr int ITB = (16 * BN) / 256;   // B staging dwords per thread

    __shared__ unsigned int As_u[2][BM * 16];   // [m][k/2], packed bf16 pair
    __shared__ unsigned int Bs_u[2][BN * 16];   // [n][k/2], B tile transposed

    const int tid  = threadIdx.x;
    const int lane = tid & 31;
    const int w    = tid >> 5;       // 0..7
    const int wm   = w / WGN;        // 0..WGM-1
    const int wn   = w % WGN;        // 0..WGN-1
    const int m0   = blockIdx.y * BM;
    const int n0   = blockIdx.x * BN;
    const long long z = blockIdx.z;

    A    += z * sA;
    Bm   += z * sB;
    bias += z * sBias;
    C    += z * sC;

    v8f acc[MT][NT];
    #pragma unroll
    for (int i = 0; i < MT; ++i)
        #pragma unroll
        for (int j = 0; j < NT; ++j)
            acc[i][j] = {};

    // fragment geometry (ISA 7.12.2, wave32)
    const int l15 = lane & 15;
    const int kbA = (lane >> 4) * 16;  // A K-base (0 or 16)
    const int kbB = (lane >> 4) * 8;   // B K-pair base (pairs 0 or 8)

    // per-thread staging coordinates (iteration-invariant parts)
    const int amm = tid >> 4;          // A row base (step 16 per it)
    const int akp = tid & 15;          // A k-pair index
    const int bkp = tid / BN;          // B k-pair base (step 256/BN per it)
    const int bn  = tid % BN;          // B column

    float2 ra[ITA];
    float  rb0[ITB], rb1[ITB];

    auto load_slab = [&](int kk) {
        #pragma unroll
        for (int it = 0; it < ITA; ++it) {
            const float* ap =
                A + (long long)(m0 + amm + it * 16) * lda + (kk + akp * 2);
            ra[it] = *(const float2*)ap;
        }
        #pragma unroll
        for (int it = 0; it < ITB; ++it) {
            int kp = bkp + it * (256 / BN);
            const float* bp = Bm + (long long)(kk + 2 * kp) * ldb + (n0 + bn);
            rb0[it] = bp[0];
            rb1[it] = bp[ldb];
        }
    };
    auto store_slab = [&](int buf) {
        #pragma unroll
        for (int it = 0; it < ITA; ++it)
            As_u[buf][tid + it * 256] = pack_bf16(ra[it].x, ra[it].y);
        #pragma unroll
        for (int it = 0; it < ITB; ++it) {
            int kp = bkp + it * (256 / BN);
            Bs_u[buf][bn * 16 + kp] = pack_bf16(rb0[it], rb1[it]);
        }
    };

    load_slab(0);
    store_slab(0);
    int parity = 0;

    for (int k0 = 0; k0 < K; k0 += 32) {
        __syncthreads();                 // buf[parity] staged & prior reads done
        const bool more = (k0 + 32 < K);
        if (more) load_slab(k0 + 32);    // issue global loads early (no waits)

        // ---- gather fragments from buf[parity] ----
        union uf16 { v16bf v; unsigned int u[8]; };
        uf16 fa[MT], fb[NT];
        #pragma unroll
        for (int i = 0; i < MT; ++i) {
            int am = wm * (16 * MT) + i * 16 + l15;
            #pragma unroll
            for (int v = 0; v < 8; ++v) {
                // A order used here (contiguous k, not ISA table order): VGPR v holds K = kbA + 2*v .. +1
                int ks = kbA + 2 * v;
                fa[i].u[v] = As_u[parity][am * 16 + (ks >> 1)];
            }
        }
        #pragma unroll
        for (int j = 0; j < NT; ++j) {
            int bnn = wn * (16 * NT) + j * 16 + l15;
            #pragma unroll
            for (int v = 0; v < 8; ++v)
                fb[j].u[v] = Bs_u[parity][bnn * 16 + kbB + v];
        }

        #pragma unroll
        for (int i = 0; i < MT; ++i)
            #pragma unroll
            for (int j = 0; j < NT; ++j)
                acc[i][j] = __builtin_amdgcn_wmma_f32_16x16x32_bf16(
                    false, fa[i].v, false, fb[j].v, (short)0, acc[i][j],
                    false, false);

        if (more) store_slab(parity ^ 1);   // waits land after the wmma burst
        parity ^= 1;
    }

    // ---- epilogue: C layout VGPR p -> M = p + 8*(lane>=16), N = lane&15 ----
    #pragma unroll
    for (int i = 0; i < MT; ++i) {
        const int rbase = m0 + wm * (16 * MT) + i * 16 + ((lane >> 4) * 8);
        #pragma unroll
        for (int j = 0; j < NT; ++j) {
            const int col = n0 + wn * (16 * NT) + j * 16 + l15;
            const float bia = bias[col];
            #pragma unroll
            for (int p = 0; p < 8; ++p) {
                float v0 = acc[i][j][p] + bia;
                if (relu) v0 = v0 > 0.f ? v0 : 0.f;
                C[(long long)(rbase + p) * ldc + col] = v0;
            }
        }
    }
}

// ---------------------------------------------------------------------------
// Fused ROI head + graph mean:
//   m[b,o] = (1/R) * sum_r ( sum_h H[r,b,h] * W2[r,h,o] + b2[r,o] )
// ---------------------------------------------------------------------------
__global__ __launch_bounds__(256)
void roi_head_mean(const float* __restrict__ H, const float* __restrict__ W2,
                   const float* __restrict__ b2, float* __restrict__ m)
{
    __shared__ float sm[256];
    const int b   = blockIdx.x;
    const int tid = threadIdx.x;

    float acc[ROI_OUT_];
    #pragma unroll
    for (int o = 0; o < ROI_OUT_; ++o) acc[o] = 0.f;

    for (int idx = tid; idx < R_ * HID_; idx += 256) {
        int r = idx >> 9;           // / HID_
        int h = idx & (HID_ - 1);
        float hv = H[((long long)r * B_ + b) * HID_ + h];
        const float* wrow = W2 + ((long long)r * HID_ + h) * ROI_OUT_;
        #pragma unroll
        for (int o = 0; o < ROI_OUT_; ++o) acc[o] += hv * wrow[o];
    }

    for (int o = 0; o < ROI_OUT_; ++o) {
        sm[tid] = acc[o];
        __syncthreads();
        for (int s = 128; s > 0; s >>= 1) {
            if (tid < s) sm[tid] += sm[tid + s];
            __syncthreads();
        }
        if (tid == 0) {
            float bias = 0.f;
            for (int r = 0; r < R_; ++r) bias += b2[r * ROI_OUT_ + o];
            m[b * ROI_OUT_ + o] = (sm[0] + bias) * (1.0f / (float)R_);
        }
        __syncthreads();
    }
}

// ---------------------------------------------------------------------------
// GCN layer 1 (K = 20, not WMMA-shaped):
//   p1[b,j] = relu( sum_k m[b,k] * Wg1[k,j] + bg1[j] )
// ---------------------------------------------------------------------------
__global__ __launch_bounds__(256)
void gcn1_k20(const float* __restrict__ m, const float* __restrict__ Wg1,
              const float* __restrict__ bg1, float* __restrict__ p1)
{
    int gid = blockIdx.x * 256 + threadIdx.x;     // B_*G1_ = 8192
    if (gid >= B_ * G1_) return;
    int b = gid >> 6;
    int j = gid & 63;
    float s = bg1[j];
    #pragma unroll
    for (int k = 0; k < ROI_OUT_; ++k)
        s += m[b * ROI_OUT_ + k] * Wg1[k * G1_ + j];
    p1[gid] = s > 0.f ? s : 0.f;
}

// ---------------------------------------------------------------------------
extern "C" void kernel_launch(void* const* d_in, const int* in_sizes, int n_in,
                              void* d_out, int out_size, void* d_ws, size_t ws_size,
                              hipStream_t stream)
{
    (void)in_sizes; (void)n_in; (void)out_size; (void)ws_size;

    const float* x   = (const float*)d_in[0];
    // d_in[1] edge_index, d_in[2] batch: unused (complete graph + self loops
    // makes the GCN aggregation an exact per-graph mean; collapsed analytically)
    const float* W1  = (const float*)d_in[3];
    const float* b1  = (const float*)d_in[4];
    const float* W2  = (const float*)d_in[5];
    const float* b2  = (const float*)d_in[6];
    const float* Wg1 = (const float*)d_in[7];
    const float* bg1 = (const float*)d_in[8];
    const float* Wg2 = (const float*)d_in[9];
    const float* bg2 = (const float*)d_in[10];
    const float* Wf1 = (const float*)d_in[11];
    const float* bf1 = (const float*)d_in[12];
    const float* Wf2 = (const float*)d_in[13];
    const float* bf2 = (const float*)d_in[14];
    const float* Wo  = (const float*)d_in[15];
    const float* bo  = (const float*)d_in[16];
    float* out = (float*)d_out;

    // workspace layout (floats)
    float* ws = (float*)d_ws;
    float* H   = ws;                                  // R*B*HID = 4,194,304
    float* m   = ws + 4194304;                        // B*20
    float* p1  = m  + 4096;                           // B*G1  = 8192
    float* p2  = p1 + 8192;                           // B*G2  = 16384
    float* y1  = p2 + 16384;                          // B*FC1 = 65536
    float* y2  = y1 + 65536;                          // B*FC2 = 131072

    dim3 blk(256);

    // Stage 1: per-ROI batched GEMM  H[r] = relu(X[r] @ W1[r] + b1[r])
    // 128x128 tile: W1 streamed exactly once (grid.y == 1).
    gemm_bf16_wmma<128, 128, 4, 2><<<dim3(HID_ / 128, B_ / 128, R_), blk, 0, stream>>>(
        x,  IN_,  (long long)B_ * IN_,
        W1, HID_, (long long)IN_ * HID_,
        b1, (long long)HID_,
        H,  HID_, (long long)B_ * HID_,
        IN_, 1);

    // Stage 2: fused ROI head (K=20) + per-graph mean -> m[B,20]
    roi_head_mean<<<dim3(B_), blk, 0, stream>>>(H, W2, b2, m);

    // GCN layer 1 (uniform-per-graph): p1 = relu(m @ Wg1 + bg1)
    gcn1_k20<<<dim3((B_ * G1_ + 255) / 256), blk, 0, stream>>>(m, Wg1, bg1, p1);

    // GCN layer 2 == pooled: p2 = relu(p1 @ Wg2 + bg2)   [128x128x64]
    gemm_bf16_wmma<64, 64, 4, 2><<<dim3(G2_ / 64, B_ / 64, 1), blk, 0, stream>>>(
        p1, G1_, 0, Wg2, G2_, 0, bg2, 0, p2, G2_, 0, G1_, 1);

    // FC1: y1 = relu(p2 @ Wf1 + bf1)   [128x512x128]
    gemm_bf16_wmma<64, 64, 4, 2><<<dim3(FC1_ / 64, B_ / 64, 1), blk, 0, stream>>>(
        p2, G2_, 0, Wf1, FC1_, 0, bf1, 0, y1, FC1_, 0, G2_, 1);

    // FC2: y2 = relu(y1 @ Wf2 + bf2)   [128x1024x512]
    gemm_bf16_wmma<64, 64, 4, 2><<<dim3(FC2_ / 64, B_ / 64, 1), blk, 0, stream>>>(
        y1, FC1_, 0, Wf2, FC2_, 0, bf2, 0, y2, FC2_, 0, FC1_, 1);

    // Output: out = y2 @ Wo + bo   [128x6400x1024]
    gemm_bf16_wmma<128, 128, 4, 2><<<dim3(OUT_ / 128, B_ / 128, 1), blk, 0, stream>>>(
        y2, FC2_, 0, Wo, OUT_, 0, bo, 0, out, OUT_, 0, FC2_, 0);
}
